// GridSelfAttention_23596550324275
// MI455X (gfx1250) — matrix operand order repaired and checked by reading, not compiled, not run on hardware
//
#include <hip/hip_runtime.h>
#include <hip/hip_bf16.h>
#include <math.h>

typedef __bf16 bf16_t;
typedef __attribute__((ext_vector_type(16))) bf16_t v16bf;
typedef __attribute__((ext_vector_type(8)))  bf16_t v8bf;
typedef __attribute__((ext_vector_type(4)))  bf16_t v4bf;
typedef __attribute__((ext_vector_type(8)))  float  v8f;
typedef __attribute__((ext_vector_type(4)))  unsigned int v4u;
typedef __attribute__((ext_vector_type(8)))  int v8i_;
typedef __attribute__((ext_vector_type(4)))  int v4i_;

#define N_DIM 384
#define C_PAIR 128
#define NUM_HEAD 4
#define QKV_DIM 32
#define NN (N_DIM * N_DIM)           // 147456
#define NT16 (N_DIM / 16)            // 24

static __device__ __forceinline__ float fast_rcp(float x) {
  return __builtin_amdgcn_rcpf(x);
}
static __device__ __forceinline__ float fast_sigmoid(float v) {
  return fast_rcp(1.0f + __expf(-v));
}
static __device__ __forceinline__ float red32_sum(float v) {
#pragma unroll
  for (int m = 1; m < 32; m <<= 1) v += __shfl_xor(v, m, 32);
  return v;
}
static __device__ __forceinline__ float red16_sum(float v) {
#pragma unroll
  for (int m = 1; m < 16; m <<= 1) v += __shfl_xor(v, m, 32);
  return v;
}
static __device__ __forceinline__ float red16_max(float v) {
#pragma unroll
  for (int m = 1; m < 16; m <<= 1) v = fmaxf(v, __shfl_xor(v, m, 32));
  return v;
}
static __device__ __forceinline__ v16bf cat8(v8bf lo, v8bf hi) {
  v16bf r;
#pragma unroll
  for (int i = 0; i < 8; ++i) { r[i] = lo[i]; r[8 + i] = hi[i]; }
  return r;
}
static __device__ __forceinline__ v8f zero8() {
  v8f z;
#pragma unroll
  for (int i = 0; i < 8; ++i) z[i] = 0.0f;
  return z;
}
static __device__ __forceinline__ v8f wmma_bf16(v16bf a, v16bf b, v8f c) {
  return __builtin_amdgcn_wmma_f32_16x16x32_bf16(false, a, false, b, (short)0, c,
                                                 false, false);
}

// Issue a TDM 2-D tile load: tile_dim0 x tile_dim1 elements of data_size=2B,
// row stride `stride0` elements, global -> LDS (contiguous). ISA ch.8 D# layout.
static __device__ __forceinline__ void tdm_load_2d_bf16(unsigned lds_off,
                                                        unsigned long long gaddr,
                                                        int tile_d0, int tile_d1,
                                                        int stride0) {
  v4u g0;
  g0[0] = 1u;                                        // count=1, user descriptor
  g0[1] = lds_off;                                   // lds_addr [63:32]
  g0[2] = (unsigned)(gaddr & 0xffffffffull);         // global_addr[31:0]
  g0[3] = (unsigned)((gaddr >> 32) & 0x1ffffffull)   // global_addr[56:32]
          | (2u << 30);                              // type=2 ("image")
  v8i_ g1;
  g1[0] = 0x00010000;                                // data_size=1 -> 2 bytes
  g1[1] = (tile_d0 & 0xffff) << 16;                  // tensor_dim0[15:0]
  g1[2] = (tile_d1 & 0xffff) << 16;                  // tensor_dim1[15:0] (dim0 hi=0)
  g1[3] = (tile_d0 & 0xffff) << 16;                  // tile_dim0 (dim1 hi=0)
  g1[4] = (tile_d1 & 0xffff);                        // tile_dim1 (tile_dim2=0)
  g1[5] = stride0;                                   // tensor_dim0_stride[31:0]
  g1[6] = 0;                                         // stride0 hi, stride1 lo
  g1[7] = 0;                                         // stride1 hi
  v4i_ z4 = {0, 0, 0, 0};
#if __clang_major__ >= 23
  v8i_ z8 = {0, 0, 0, 0, 0, 0, 0, 0};
  __builtin_amdgcn_tensor_load_to_lds(g0, g1, z4, z4, z8, 0);
#else
  __builtin_amdgcn_tensor_load_to_lds(g0, g1, z4, z4, 0);
#endif
}

// ---------------- Kernel 0: convert 5 weight matrices f32 -> bf16 -------------
__global__ void k_wcvt(const float* __restrict__ wq, const float* __restrict__ wk,
                       const float* __restrict__ wv, const float* __restrict__ wg,
                       const float* __restrict__ wo, bf16_t* __restrict__ wbf) {
  int idx = blockIdx.x * blockDim.x + threadIdx.x;  // 5*16384 total
  int which = idx >> 14;
  int off = idx & 16383;
  const float* src = (which == 0) ? wq : (which == 1) ? wk : (which == 2) ? wv
                    : (which == 3) ? wg : wo;
  wbf[idx] = (bf16_t)src[off];
}

// -------- Kernel 1: LayerNorm (per pair-row) + per-head bias projection -------
__global__ void k_ln(const float* __restrict__ pair, const float* __restrict__ lnw,
                     const float* __restrict__ lnb, const float* __restrict__ wbias,
                     bf16_t* __restrict__ xbf, float* __restrict__ biasf) {
  int wave = threadIdx.x >> 5;
  int lane = threadIdx.x & 31;
  size_t p = (size_t)blockIdx.x * 8 + wave;  // pair row index, < NN
  const float4 xv = *(const float4*)(pair + p * C_PAIR + lane * 4);
  float s = xv.x + xv.y + xv.z + xv.w;
  float ss = xv.x * xv.x + xv.y * xv.y + xv.z * xv.z + xv.w * xv.w;
  s = red32_sum(s);
  ss = red32_sum(ss);
  float mu = s * (1.0f / C_PAIR);
  float var = ss * (1.0f / C_PAIR) - mu * mu;
  float inv = __builtin_amdgcn_rsqf(var + 1e-5f);
  const float4 w4 = *(const float4*)(lnw + lane * 4);
  const float4 b4 = *(const float4*)(lnb + lane * 4);
  float4 xn;
  xn.x = (xv.x - mu) * inv * w4.x + b4.x;
  xn.y = (xv.y - mu) * inv * w4.y + b4.y;
  xn.z = (xv.z - mu) * inv * w4.z + b4.z;
  xn.w = (xv.w - mu) * inv * w4.w + b4.w;
  v4bf o;
  o[0] = (bf16_t)xn.x; o[1] = (bf16_t)xn.y; o[2] = (bf16_t)xn.z; o[3] = (bf16_t)xn.w;
  *(v4bf*)(xbf + p * C_PAIR + lane * 4) = o;
#pragma unroll
  for (int h = 0; h < NUM_HEAD; ++h) {
    const float4 wb = *(const float4*)(wbias + h * C_PAIR + lane * 4);
    float t = xn.x * wb.x + xn.y * wb.y + xn.z * wb.z + xn.w * wb.w;
    t = red32_sum(t);
    if (lane == 0) biasf[(size_t)h * NN + p] = t;
  }
}

// ---- Kernel 2: Q/K/V/gate projections via WMMA (y = x W^T), bf16 outputs ----
__global__ __launch_bounds__(256) void k_proj(const bf16_t* __restrict__ xbf,
                                              const bf16_t* __restrict__ wbf,
                                              bf16_t* __restrict__ qbf,
                                              bf16_t* __restrict__ kbf,
                                              bf16_t* __restrict__ vbf,
                                              bf16_t* __restrict__ gbf) {
  int wave = threadIdx.x >> 5;
  int lane = threadIdx.x & 31;
  int m = lane & 15;
  int half = lane >> 4;
  int kb = half * 8;
  int tile = blockIdx.x * 8 + wave;  // < NN/16
  size_t row0 = (size_t)tile * 16;

  v16bf af[4];
#pragma unroll
  for (int ks = 0; ks < 4; ++ks) {
    const bf16_t* a = xbf + (row0 + m) * C_PAIR + ks * 32 + kb;
    af[ks] = cat8(*(const v8bf*)a, *(const v8bf*)(a + 16));
  }
#pragma unroll
  for (int mat = 0; mat < 4; ++mat) {
    const bf16_t* W = wbf + (size_t)mat * C_PAIR * C_PAIR;
    bf16_t* dst = (mat == 0) ? qbf : (mat == 1) ? kbf : (mat == 2) ? vbf : gbf;
    for (int ot = 0; ot < 8; ++ot) {
      v16bf bfr[4];
#pragma unroll
      for (int ks = 0; ks < 4; ++ks)
        bfr[ks] = cat8(*(const v8bf*)(W + (size_t)(ot * 16 + m) * C_PAIR + ks * 32 + kb), *(const v8bf*)(W + (size_t)(ot * 16 + m) * C_PAIR + ks * 32 + 16 + kb));
      v8f acc = zero8();
#pragma unroll
      for (int ks = 0; ks < 4; ++ks) acc = wmma_bf16(af[ks], bfr[ks], acc);
      bf16_t* drow = dst + (row0 + half * 8) * C_PAIR + ot * 16 + m;
#pragma unroll
      for (int r = 0; r < 8; ++r) {
        float v = acc[r];
        if (mat == 0) v *= 0.17677669529663687f;  // 1/sqrt(32) folded into Q
        if (mat == 3) v = fast_sigmoid(v);        // gate
        drow[(size_t)r * C_PAIR] = (bf16_t)v;
      }
    }
  }
}

// --------------- Kernel 3: attention per (b, h), fully fused -----------------
__global__ __launch_bounds__(128) void k_attn(const bf16_t* __restrict__ qbf,
                                              const bf16_t* __restrict__ kbf,
                                              const bf16_t* __restrict__ vbf,
                                              const float* __restrict__ biasf,
                                              const int* __restrict__ maskp,
                                              bf16_t* __restrict__ wabf) {
  __shared__ __align__(32) bf16_t Vrows[N_DIM][QKV_DIM];    // 24 KB, TDM landing pad
  __shared__ __align__(32) bf16_t Vt[QKV_DIM][N_DIM];       // 24 KB, V transposed
  __shared__ __align__(32) bf16_t Pbf[4][16][N_DIM];        // 48 KB, per-wave probs
  __shared__ float msk[N_DIM];                              // 1.5 KB

  const int b = blockIdx.x;
  const int h = blockIdx.y;
  const int tid = threadIdx.x;
  const int wave = tid >> 5;
  const int lane = tid & 31;
  const int m = lane & 15;
  const int half = lane >> 4;
  const int kb = half * 8;

  // TDM: DMA the V tile [384 x 32] bf16 (row stride 128 elems) into LDS.
  // Only wave 0 reaches the instruction; TDM ignores EXEC, so it issues once.
  if (tid == 0) {
    tdm_load_2d_bf16((unsigned)(uintptr_t)&Vrows[0][0],
                     (unsigned long long)(uintptr_t)(vbf +
                         ((size_t)b * N_DIM) * C_PAIR + h * QKV_DIM),
                     QKV_DIM, N_DIM, C_PAIR);
  }

  // warm WGP$/L2 with the K rows (each row is re-read 24x by this block)
  for (int j = tid; j < N_DIM; j += 128)
    __builtin_prefetch(kbf + ((size_t)b * N_DIM + j) * C_PAIR + h * QKV_DIM, 0, 3);
  for (int j = tid; j < N_DIM; j += 128)
    msk[j] = maskp[b * N_DIM + j] ? 0.0f : -1e9f;

  __builtin_amdgcn_s_wait_tensorcnt(0);  // wave0: tile landed; others: no-op
  __syncthreads();

  // transpose V in LDS: Vt[d][j] = Vrows[j][d] (vector reads, b16 scatter writes)
  for (int c = tid; c < N_DIM * QKV_DIM / 8; c += 128) {
    int j = c >> 2;
    int d0 = (c & 3) * 8;
    v8bf r = *(const v8bf*)&Vrows[j][d0];
#pragma unroll
    for (int i = 0; i < 8; ++i) Vt[d0 + i][j] = r[i];
  }
  __syncthreads();

  for (int c = wave; c < NT16; c += 4) {
    const int i0 = c * 16;
    // Q A-fragment straight from global (d == K-dim == 32 exactly)
    const bf16_t* qrow = qbf + ((size_t)b * N_DIM + i0 + m) * C_PAIR + h * QKV_DIM;
    v16bf aq = cat8(*(const v8bf*)(qrow + kb), *(const v8bf*)(qrow + 16 + kb));

    v8f L[NT16];
#pragma unroll
    for (int t = 0; t < NT16; ++t) {
      const int j0 = t * 16;
      // K rows are contiguous in d -> B fragment is two 16B loads (table k order)
      const bf16_t* krow = kbf + ((size_t)b * N_DIM + j0 + m) * C_PAIR + h * QKV_DIM;
      v16bf bk = cat8(*(const v8bf*)(krow + kb), *(const v8bf*)(krow + 16 + kb));
      v8f z = wmma_bf16(aq, bk, zero8());
      const float mv = msk[j0 + m];
      const float* bp = biasf + (size_t)h * NN + (size_t)(i0 + half * 8) * N_DIM + j0 + m;
#pragma unroll
      for (int r = 0; r < 8; ++r) z[r] += bp[(size_t)r * N_DIM] + mv;
      L[t] = z;
    }

    // softmax along j: row M = r + 8*half lives in one 16-lane group
    float rinv[8];
#pragma unroll
    for (int r = 0; r < 8; ++r) {
      float mx = -3.0e38f;
#pragma unroll
      for (int t = 0; t < NT16; ++t) mx = fmaxf(mx, L[t][r]);
      mx = red16_max(mx);
      float s = 0.0f;
#pragma unroll
      for (int t = 0; t < NT16; ++t) {
        float e = __expf(L[t][r] - mx);
        L[t][r] = e;
        s += e;
      }
      s = red16_sum(s);
      rinv[r] = fast_rcp(s);
    }
    // probs -> bf16 in A-layout staging buffer (wave-private, LDS is in-order)
#pragma unroll
    for (int t = 0; t < NT16; ++t)
#pragma unroll
      for (int r = 0; r < 8; ++r)
        Pbf[wave][r + half * 8][t * 16 + m] = (bf16_t)(L[t][r] * rinv[r]);

    // wa = P x V, K-dim = 384 = 12 steps of 32
#pragma unroll
    for (int nt = 0; nt < 2; ++nt) {
      v8f acc = zero8();
#pragma unroll
      for (int s = 0; s < 12; ++s) {
        const bf16_t* pp = &Pbf[wave][m][s * 32 + kb];
        v16bf ap = cat8(*(const v8bf*)pp, *(const v8bf*)(pp + 16));
        v16bf bv = cat8(*(const v8bf*)&Vt[nt * 16 + m][s * 32 + kb], *(const v8bf*)&Vt[nt * 16 + m][s * 32 + 16 + kb]);
        acc = wmma_bf16(ap, bv, acc);
      }
      bf16_t* wrow = wabf + ((size_t)b * N_DIM + i0 + half * 8) * C_PAIR +
                     h * QKV_DIM + nt * 16 + m;
#pragma unroll
      for (int r = 0; r < 8; ++r) wrow[(size_t)r * C_PAIR] = (bf16_t)acc[r];
    }
  }
}

// ------------- Kernel 4: out = (wa * gate) @ w_out^T, f32 output -------------
__global__ __launch_bounds__(256) void k_out(const bf16_t* __restrict__ wabf,
                                             const bf16_t* __restrict__ gbf,
                                             const bf16_t* __restrict__ wob,
                                             float* __restrict__ out) {
  int wave = threadIdx.x >> 5;
  int lane = threadIdx.x & 31;
  int m = lane & 15;
  int half = lane >> 4;
  int kb = half * 8;
  int tile = blockIdx.x * 8 + wave;
  size_t row0 = (size_t)tile * 16;

  v16bf af[4];
#pragma unroll
  for (int ks = 0; ks < 4; ++ks) {
    size_t base = (row0 + m) * C_PAIR + ks * 32 + kb;
    v8bf wl = *(const v8bf*)(wabf + base);
    v8bf wh = *(const v8bf*)(wabf + base + 16);
    v8bf gl = *(const v8bf*)(gbf + base);
    v8bf gh = *(const v8bf*)(gbf + base + 16);
    v16bf a;
#pragma unroll
    for (int i = 0; i < 8; ++i) {
      a[i] = (bf16_t)((float)wl[i] * (float)gl[i]);
      a[8 + i] = (bf16_t)((float)wh[i] * (float)gh[i]);
    }
    af[ks] = a;
  }
  for (int ot = 0; ot < 8; ++ot) {
    v16bf bfr[4];
#pragma unroll
    for (int ks = 0; ks < 4; ++ks)
      bfr[ks] = cat8(*(const v8bf*)(wob + (size_t)(ot * 16 + m) * C_PAIR + ks * 32 + kb), *(const v8bf*)(wob + (size_t)(ot * 16 + m) * C_PAIR + ks * 32 + 16 + kb));
    v8f acc = zero8();
#pragma unroll
    for (int ks = 0; ks < 4; ++ks) acc = wmma_bf16(af[ks], bfr[ks], acc);
    float* orow = out + (row0 + half * 8) * C_PAIR + ot * 16 + m;
#pragma unroll
    for (int r = 0; r < 8; ++r) orow[(size_t)r * C_PAIR] = acc[r];
  }
}

extern "C" void kernel_launch(void* const* d_in, const int* in_sizes, int n_in,
                              void* d_out, int out_size, void* d_ws, size_t ws_size,
                              hipStream_t stream) {
  const float* pair   = (const float*)d_in[0];
  const int*   mask   = (const int*)d_in[1];
  const float* ln_w   = (const float*)d_in[2];
  const float* ln_b   = (const float*)d_in[3];
  const float* w_bias = (const float*)d_in[4];
  const float* w_q    = (const float*)d_in[5];
  const float* w_k    = (const float*)d_in[6];
  const float* w_v    = (const float*)d_in[7];
  const float* w_gate = (const float*)d_in[8];
  const float* w_out  = (const float*)d_in[9];
  float* out = (float*)d_out;

  char* ws = (char*)d_ws;
  const size_t nnc_bf = (size_t)NN * C_PAIR * sizeof(bf16_t);  // 37.75 MB
  size_t off = 0;
  bf16_t* xbf  = (bf16_t*)(ws + off); off += nnc_bf;
  bf16_t* qbf  = (bf16_t*)(ws + off); off += nnc_bf;
  bf16_t* kbf  = (bf16_t*)(ws + off); off += nnc_bf;
  bf16_t* vbf  = (bf16_t*)(ws + off); off += nnc_bf;
  bf16_t* gbf  = (bf16_t*)(ws + off); off += nnc_bf;
  bf16_t* wabf = (bf16_t*)(ws + off); off += nnc_bf;
  float*  biasf = (float*)(ws + off); off += (size_t)NUM_HEAD * NN * sizeof(float);
  bf16_t* wbf  = (bf16_t*)(ws + off); off += 5 * (size_t)C_PAIR * C_PAIR * sizeof(bf16_t);

  k_wcvt<<<320, 256, 0, stream>>>(w_q, w_k, w_v, w_gate, w_out, wbf);
  k_ln<<<NN / 8, 256, 0, stream>>>(pair, ln_w, ln_b, w_bias, xbf, biasf);
  k_proj<<<NN / 16 / 8, 256, 0, stream>>>(xbf, wbf, qbf, kbf, vbf, gbf);
  dim3 g3(N_DIM, NUM_HEAD);
  k_attn<<<g3, 128, 0, stream>>>(qbf, kbf, vbf, biasf, mask, wabf);
  k_out<<<NN / 16 / 8, 256, 0, stream>>>(wabf, gbf, wbf + 4 * C_PAIR * C_PAIR, out);
}
